// DecayAware_73169062854706
// MI455X (gfx1250) — compile-verified, matrix operand order checked by reading, not run on hardware
//
#include <hip/hip_runtime.h>
#include <hip/hip_bf16.h>

// ---------------- constants ----------------
#define BB   512
#define TT   256
#define DD   128
#define HH   256
#define OO   128
#define D3   384
#define D3H  640
#define BT   131072   // B*T

// bf16 weight region, element offsets
#define WEMB_E 0
#define WD1_E  16384
#define WD2_E  32768
#define WC_E   49152
#define WZX_E  81920
#define WRX_E  180224
#define WHX_E  278528
#define WZH_E  376832
#define WRH_E  442368
#define WHH_E  507904
#define WO_E   573440
#define WBF_TOTAL 606208

// workspace byte offsets
#define ZX_B     0u
#define RX_B     67108864u
#define HX_B     134217728u
#define XBAR_B   201326592u
#define XMEAN_B  201588736u
#define HLAST_B  201850880u
#define WBF_B    202375168u

typedef __attribute__((ext_vector_type(16))) __bf16 v16bf;
typedef __attribute__((ext_vector_type(8)))  float  v8f;

struct alignas(16) U128 { unsigned a, b, c, d; };

// ---------------- helpers ----------------
__device__ inline float n2n(float v) { return __builtin_isfinite(v) ? v : 0.f; }
__device__ inline float sigmoidf_(float x) { return 1.f / (1.f + __expf(-x)); }

__device__ inline __bf16 f2bf(float f) {
  unsigned u = __builtin_bit_cast(unsigned, f);
  unsigned r = u + 0x7FFFu + ((u >> 16) & 1u);      // RNE truncate
  unsigned short s = (unsigned short)(r >> 16);
  return __builtin_bit_cast(__bf16, s);
}
__device__ inline float bf2f(__bf16 b) {
  unsigned u = ((unsigned)__builtin_bit_cast(unsigned short, b)) << 16;
  return __builtin_bit_cast(float, u);
}

// Per-lane async copy: 16 bytes global -> LDS, tracked by ASYNCcnt.
__device__ inline void async_ld16(unsigned lds_addr, const void* gptr) {
  unsigned long long ga = (unsigned long long)(uintptr_t)gptr;
  asm volatile("global_load_async_to_lds_b128 %0, %1, off"
               :: "v"(lds_addr), "v"(ga) : "memory");
}
__device__ inline void wait_async0() {
  asm volatile("s_wait_asynccnt 0x0" ::: "memory");
}

// 16x32 bf16 A/B fragment load from row-major storage (LDS or global).
// Lane L (row = L&15) loads two 16B chunks at k = kbase + 8*(L>>4) and +16.
__device__ inline v16bf load_frag(const __bf16* p, int ld, int row0, int kbase) {
  int lane = threadIdx.x & 31;
  int lo = lane & 15, hi = lane >> 4;
  const __bf16* q = p + (size_t)(row0 + lo) * ld + kbase + hi * 8;
  union { v16bf v; U128 u[2]; } x;
  x.u[0] = *(const U128*)(q);
  x.u[1] = *(const U128*)(q + 16);
  return x.v;
}

__device__ inline v8f wmma_bf(v16bf a, v16bf b, v8f c) {
  return __builtin_amdgcn_wmma_f32_16x16x32_bf16(false, a, false, b, (short)0, c,
                                                 false, false);
}

// ---------------- kernel: weight fp32 -> bf16 repack ----------------
__global__ void __launch_bounds__(256) k_wconv(
    const float* W_emb, const float* W_d1, const float* W_d2, const float* W_c,
    const float* W_z, const float* W_r, const float* W_h, const float* W_o,
    __bf16* wbf) {
  int i = blockIdx.x * 256 + threadIdx.x;
  if (i >= WBF_TOTAL) return;
  float v;
  if      (i < WD1_E)  v = W_emb[i];
  else if (i < WD2_E)  v = W_d1[i - WD1_E];
  else if (i < WC_E)   v = W_d2[i - WD2_E];
  else if (i < WZX_E)  v = W_c[i - WC_E];
  else if (i < WRX_E) { int j = i - WZX_E; v = W_z[(j / D3) * D3H + (j % D3)]; }
  else if (i < WHX_E) { int j = i - WRX_E; v = W_r[(j / D3) * D3H + (j % D3)]; }
  else if (i < WZH_E) { int j = i - WHX_E; v = W_h[(j / D3) * D3H + (j % D3)]; }
  else if (i < WRH_E) { int j = i - WZH_E; v = W_z[(j / HH) * D3H + D3 + (j % HH)]; }
  else if (i < WHH_E) { int j = i - WRH_E; v = W_r[(j / HH) * D3H + D3 + (j % HH)]; }
  else if (i < WO_E)  { int j = i - WHH_E; v = W_h[(j / HH) * D3H + D3 + (j % HH)]; }
  else                 v = W_o[i - WO_E];
  wbf[i] = f2bf(v);
}

// ---------------- kernel: xbar = mean_t(x) ----------------
__global__ void __launch_bounds__(256) k_xbar(const float* x, float* xbar) {
  int i = blockIdx.x * 256 + threadIdx.x;   // B*D threads
  int b = i >> 7, c = i & 127;
  const float* p = x + (size_t)b * TT * DD + c;
  float s = 0.f;
  for (int t = 0; t < TT; ++t) s += p[t * DD];
  xbar[i] = s * (1.f / TT);
}

// ---------------- kernel: x_mean = n2n(xbar @ Wemb^T + b) ----------------
__global__ void __launch_bounds__(256) k_xmean(const float* xbar, const __bf16* wbf,
                                               const float* b_emb, float* xmean) {
  __shared__ __align__(16) __bf16 sA[16 * 128];
  int r0 = blockIdx.x * 16;
  for (int i = threadIdx.x; i < 16 * 128; i += 256)
    sA[i] = f2bf(xbar[r0 * 128 + i]);
  __syncthreads();
  int w = threadIdx.x >> 5, lane = threadIdx.x & 31;
  int lo = lane & 15, hi = lane >> 4, n0 = w * 16;
  v8f acc = {};
#pragma unroll
  for (int kc = 0; kc < 4; ++kc)
    acc = wmma_bf(load_frag(sA, 128, 0, kc * 32),
                  load_frag(wbf + WEMB_E, 128, n0, kc * 32), acc);
  int col = n0 + lo;
  float bb = b_emb[col];
#pragma unroll
  for (int r = 0; r < 8; ++r)
    xmean[(r0 + r + 8 * hi) * DD + col] = n2n(acc[r] + bb);
}

// ---------------- kernel: fused feature pipeline + gate projections ----------
__global__ void __launch_bounds__(256) k_features(
    const float* x, const float* x_mask, const float* x_delta, const float* noise,
    const float* b_emb, const float* b_d1, const float* b_d2, const float* b_c,
    const float* b_z, const float* b_r, const float* b_h,
    const float* xmean, const __bf16* wbf,
    __bf16* zx, __bf16* rx, __bf16* hx) {
  __shared__ __align__(16) __bf16 bfA[16 * 384];  // A staging: [xf|m|d] finally
  __shared__ float fe[16 * 128];                  // e, then x_hat (fp32)
  __shared__ float fg[16 * 128];                  // gamma, then conf
  __shared__ float fd[16 * 128];                  // clipped delta (fp32)
  int tid = threadIdx.x;
  size_t row0 = (size_t)blockIdx.x * 16;          // row in [0, BT)
  int b = (int)(row0 >> 8);                       // T == 256

  // ---- stage 1: load x, d (clip), keep mask in registers
  float m_reg[8];
  for (int i = tid, j = 0; i < 2048; i += 256, ++j) {
    int r = i >> 7, c = i & 127;
    size_t g = (row0 + r) * DD + c;
    float xv = x[g];
    float dv = x_delta[g];
    dv = (dv != dv) ? 0.f : dv;
    dv = fminf(fmaxf(dv, 0.f), 100.f);
    bfA[r * 384 + c]       = f2bf(xv);   // x in cols [0,128)
    bfA[r * 384 + 128 + c] = f2bf(dv);   // d in cols [128,256)
    fd[i] = dv;
    m_reg[j] = x_mask[g];
  }
  __syncthreads();

  int w = tid >> 5, lane = tid & 31, lo = lane & 15, hi = lane >> 4;
  int n0 = w * 16;

  // ---- stage 2: e = n2n(x@Wemb^T+b);  t1 = relu(d@Wd1^T+b) -> bfA[256:384)
  {
    v8f acc = {};
#pragma unroll
    for (int kc = 0; kc < 4; ++kc)
      acc = wmma_bf(load_frag(bfA, 384, 0, kc * 32),
                    load_frag(wbf + WEMB_E, 128, n0, kc * 32), acc);
    int col = n0 + lo;
    float bb = b_emb[col];
#pragma unroll
    for (int r = 0; r < 8; ++r) fe[(r + 8 * hi) * 128 + col] = n2n(acc[r] + bb);
  }
  {
    v8f acc = {};
#pragma unroll
    for (int kc = 0; kc < 4; ++kc)
      acc = wmma_bf(load_frag(bfA, 384, 0, 128 + kc * 32),
                    load_frag(wbf + WD1_E, 128, n0, kc * 32), acc);
    int col = n0 + lo;
    float bb = b_d1[col];
#pragma unroll
    for (int r = 0; r < 8; ++r) {
      float t1 = fmaxf(acc[r] + bb, 0.f);
      bfA[(r + 8 * hi) * 384 + 256 + col] = f2bf(t1);
    }
  }
  __syncthreads();

  // ---- stage 3: gamma = sigmoid(t1@Wd2^T+b)
  {
    v8f acc = {};
#pragma unroll
    for (int kc = 0; kc < 4; ++kc)
      acc = wmma_bf(load_frag(bfA, 384, 0, 256 + kc * 32),
                    load_frag(wbf + WD2_E, 128, n0, kc * 32), acc);
    int col = n0 + lo;
    float bb = b_d2[col];
#pragma unroll
    for (int r = 0; r < 8; ++r)
      fg[(r + 8 * hi) * 128 + col] = sigmoidf_(acc[r] + bb);
  }
  __syncthreads();

  // ---- stage 4: x_hat = m*e + (1-m)*gamma*x_mean  -> fe (f32) and bfA[0:128)
  for (int i = tid, j = 0; i < 2048; i += 256, ++j) {
    int r = i >> 7, c = i & 127;
    float m = m_reg[j];
    float xh = m * fe[i] + (1.f - m) * (fg[i] * xmean[b * DD + c]);
    fe[i] = xh;
    bfA[r * 384 + c] = f2bf(xh);
  }
  __syncthreads();

  // ---- stage 5: conf = sigmoid([x_hat,d]@Wc^T+b)  (K=256)
  {
    v8f acc = {};
#pragma unroll
    for (int kc = 0; kc < 8; ++kc)
      acc = wmma_bf(load_frag(bfA, 384, 0, kc * 32),
                    load_frag(wbf + WC_E, 256, n0, kc * 32), acc);
    int col = n0 + lo;
    float bb = b_c[col];
#pragma unroll
    for (int r = 0; r < 8; ++r)
      fg[(r + 8 * hi) * 128 + col] = sigmoidf_(acc[r] + bb);
  }
  __syncthreads();

  // ---- stage 6: x_final; build xc = [x_final, m, d] in bfA
  for (int i = tid, j = 0; i < 2048; i += 256, ++j) {
    int r = i >> 7, c = i & 127;
    size_t g = (row0 + r) * DD + c;
    float cf = fg[i];
    float xf = n2n(fe[i] * cf + noise[g] * (1.f - cf));
    bfA[r * 384 + c]       = f2bf(xf);
    bfA[r * 384 + 128 + c] = f2bf(m_reg[j]);
    bfA[r * 384 + 256 + c] = f2bf(fd[i]);
  }
  __syncthreads();

  // ---- stage 7: zx/rx/hx = xc @ W{z,r,h}x^T + b  (3 gates x 16 col tiles)
  for (int jt = w; jt < 48; jt += 8) {
    int gate = jt >> 4;
    int nn0 = (jt & 15) * 16;
    const __bf16* Wg = wbf + (gate == 0 ? WZX_E : gate == 1 ? WRX_E : WHX_E);
    const float* bias = gate == 0 ? b_z : gate == 1 ? b_r : b_h;
    __bf16* dst = gate == 0 ? zx : gate == 1 ? rx : hx;
    __builtin_prefetch(Wg + (size_t)nn0 * D3, 0, 1);
    v8f acc = {};
#pragma unroll
    for (int kc = 0; kc < 12; ++kc)
      acc = wmma_bf(load_frag(bfA, 384, 0, kc * 32),
                    load_frag(Wg, 384, nn0, kc * 32), acc);
    int col = nn0 + lo;
    float bb = bias[col];
#pragma unroll
    for (int r = 0; r < 8; ++r)
      dst[(row0 + r + 8 * hi) * HH + col] = f2bf(acc[r] + bb);
  }
}

// ---------------- kernel: sequential GRU over T ----------------
// h resident in LDS; next step's zx/rx/hx tiles double-buffered into LDS with
// global_load_async_to_lds_b128 (ASYNCcnt), overlapped with the WMMA chains.
__global__ void __launch_bounds__(512) k_gru(const __bf16* zx, const __bf16* rx,
                                             const __bf16* hx, const __bf16* wbf,
                                             float* hlast) {
  __shared__ float hF[16 * 256];
  __shared__ __align__(16) __bf16 hB[16 * 256];
  __shared__ __align__(16) __bf16 rhB[16 * 256];
  __shared__ float zS[16 * 256];
  __shared__ __align__(16) __bf16 gbuf[2][3][16 * 256];  // [parity][gate]
  int tid = threadIdx.x;
  int b0 = blockIdx.x * 16;
  for (int i = tid; i < 4096; i += 512) { hF[i] = 0.f; hB[i] = f2bf(0.f); }

  // one 16B async transfer per lane per gate: 512 threads cover 16x256 bf16
  int prow = tid >> 5;          // 0..15
  int pcol = (tid & 31) * 8;    // 0..248 step 8
  const __bf16* gsrc0 = zx;
  const __bf16* gsrc1 = rx;
  const __bf16* gsrc2 = hx;
  auto prefetch = [&](int t, int p) {
    size_t gb = ((size_t)(b0 + prow) * TT + t) * HH + pcol;
    unsigned loff = (unsigned)(prow * 256 + pcol);
    async_ld16((unsigned)(uintptr_t)&gbuf[p][0][loff], gsrc0 + gb);
    async_ld16((unsigned)(uintptr_t)&gbuf[p][1][loff], gsrc1 + gb);
    async_ld16((unsigned)(uintptr_t)&gbuf[p][2][loff], gsrc2 + gb);
  };
  prefetch(0, 0);

  int w = tid >> 5, lane = tid & 31, lo = lane & 15, hi = lane >> 4;
  const __bf16* Wzh = wbf + WZH_E;
  const __bf16* Wrh = wbf + WRH_E;
  const __bf16* Whh = wbf + WHH_E;
  int n0 = w * 16, col = n0 + lo;

  for (int t = 0; t < TT; ++t) {
    int p = t & 1;
    wait_async0();        // own async batch for step t complete
    __syncthreads();      // publish gbuf[p] (and hB from previous phase 2)
    if (t + 1 < TT) prefetch(t + 1, p ^ 1);

    // phase 1: wave w computes z-tile(w) and r-tile(w), K = 256
    {
      v8f acc = {};
#pragma unroll
      for (int kc = 0; kc < 8; ++kc)
        acc = wmma_bf(load_frag(hB, 256, 0, kc * 32),
                      load_frag(Wzh, 256, n0, kc * 32), acc);
#pragma unroll
      for (int r = 0; r < 8; ++r) {
        int row = r + 8 * hi;
        zS[row * 256 + col] = sigmoidf_(acc[r] + bf2f(gbuf[p][0][row * 256 + col]));
      }
    }
    {
      v8f acc = {};
#pragma unroll
      for (int kc = 0; kc < 8; ++kc)
        acc = wmma_bf(load_frag(hB, 256, 0, kc * 32),
                      load_frag(Wrh, 256, n0, kc * 32), acc);
#pragma unroll
      for (int r = 0; r < 8; ++r) {
        int row = r + 8 * hi;
        float rv = sigmoidf_(acc[r] + bf2f(gbuf[p][1][row * 256 + col]));
        rhB[row * 256 + col] = f2bf(rv * hF[row * 256 + col]);
      }
    }
    __syncthreads();
    // phase 2: h_tilde = tanh(hx + (r*h)@Whh^T); h = (1-z)h + z*h_tilde
    {
      v8f acc = {};
#pragma unroll
      for (int kc = 0; kc < 8; ++kc)
        acc = wmma_bf(load_frag(rhB, 256, 0, kc * 32),
                      load_frag(Whh, 256, n0, kc * 32), acc);
#pragma unroll
      for (int r = 0; r < 8; ++r) {
        int row = r + 8 * hi;
        float ht = tanhf(acc[r] + bf2f(gbuf[p][2][row * 256 + col]));
        float z = zS[row * 256 + col];
        float hn = n2n((1.f - z) * hF[row * 256 + col] + z * ht);
        hF[row * 256 + col] = hn;
        hB[row * 256 + col] = f2bf(hn);
      }
    }
  }
  __syncthreads();
  for (int i = tid; i < 4096; i += 512)
    hlast[(size_t)(b0 + (i >> 8)) * HH + (i & 255)] = hF[i];
}

// ---------------- kernel: layernorm + output projection ----------------
__global__ void __launch_bounds__(256) k_out(const float* hlast, const float* ln_g,
                                             const float* ln_b, const float* b_o,
                                             const __bf16* wbf, float* out) {
  __shared__ float hS[16 * 256];
  __shared__ __align__(16) __bf16 lnB[16 * 256];
  __shared__ float muS[16], rsS[16];
  int tid = threadIdx.x;
  int b0 = blockIdx.x * 16;
  for (int i = tid; i < 4096; i += 256)
    hS[i] = hlast[(size_t)(b0 + (i >> 8)) * HH + (i & 255)];
  __syncthreads();
  if (tid < 16) {
    float s = 0.f;
    for (int c = 0; c < HH; ++c) s += hS[tid * 256 + c];
    float mu = s / HH;
    float v = 0.f;
    for (int c = 0; c < HH; ++c) { float d = hS[tid * 256 + c] - mu; v += d * d; }
    muS[tid] = mu;
    rsS[tid] = 1.f / sqrtf(v / HH + 1e-5f);
  }
  __syncthreads();
  for (int i = tid; i < 4096; i += 256) {
    int r = i >> 8, c = i & 255;
    lnB[i] = f2bf((hS[i] - muS[r]) * rsS[r] * ln_g[c] + ln_b[c]);
  }
  __syncthreads();
  int w = tid >> 5, lane = tid & 31, lo = lane & 15, hi = lane >> 4;
  int n0 = w * 16;
  v8f acc = {};
#pragma unroll
  for (int kc = 0; kc < 8; ++kc)
    acc = wmma_bf(load_frag(lnB, 256, 0, kc * 32),
                  load_frag(wbf + WO_E, 256, n0, kc * 32), acc);
  int col = n0 + lo;
  float bb = b_o[col];
#pragma unroll
  for (int r = 0; r < 8; ++r)
    out[(size_t)(b0 + r + 8 * hi) * OO + col] = acc[r] + bb;
}

// ---------------- launch ----------------
extern "C" void kernel_launch(void* const* d_in, const int* in_sizes, int n_in,
                              void* d_out, int out_size, void* d_ws, size_t ws_size,
                              hipStream_t stream) {
  const float* x       = (const float*)d_in[0];
  const float* x_mask  = (const float*)d_in[1];
  const float* x_delta = (const float*)d_in[2];
  const float* noise   = (const float*)d_in[3];
  const float* W_emb = (const float*)d_in[4];   const float* b_emb = (const float*)d_in[5];
  const float* W_d1  = (const float*)d_in[6];   const float* b_d1  = (const float*)d_in[7];
  const float* W_d2  = (const float*)d_in[8];   const float* b_d2  = (const float*)d_in[9];
  const float* W_c   = (const float*)d_in[10];  const float* b_c   = (const float*)d_in[11];
  const float* W_z   = (const float*)d_in[12];  const float* b_z   = (const float*)d_in[13];
  const float* W_r   = (const float*)d_in[14];  const float* b_r   = (const float*)d_in[15];
  const float* W_h   = (const float*)d_in[16];  const float* b_h   = (const float*)d_in[17];
  const float* ln_g  = (const float*)d_in[18];  const float* ln_b  = (const float*)d_in[19];
  const float* W_o   = (const float*)d_in[20];  const float* b_o   = (const float*)d_in[21];

  char* ws = (char*)d_ws;
  __bf16* zx    = (__bf16*)(ws + ZX_B);
  __bf16* rx    = (__bf16*)(ws + RX_B);
  __bf16* hx    = (__bf16*)(ws + HX_B);
  float*  xbar  = (float*)(ws + XBAR_B);
  float*  xmean = (float*)(ws + XMEAN_B);
  float*  hlast = (float*)(ws + HLAST_B);
  __bf16* wbf   = (__bf16*)(ws + WBF_B);
  float*  out   = (float*)d_out;

  k_wconv<<<dim3((WBF_TOTAL + 255) / 256), dim3(256), 0, stream>>>(
      W_emb, W_d1, W_d2, W_c, W_z, W_r, W_h, W_o, wbf);
  k_xbar<<<dim3((BB * DD) / 256), dim3(256), 0, stream>>>(x, xbar);
  k_xmean<<<dim3(BB / 16), dim3(256), 0, stream>>>(xbar, wbf, b_emb, xmean);
  k_features<<<dim3(BT / 16), dim3(256), 0, stream>>>(
      x, x_mask, x_delta, noise, b_emb, b_d1, b_d2, b_c, b_z, b_r, b_h,
      xmean, wbf, zx, rx, hx);
  k_gru<<<dim3(BB / 16), dim3(512), 0, stream>>>(zx, rx, hx, wbf, hlast);
  k_out<<<dim3(BB / 16), dim3(256), 0, stream>>>(hlast, ln_g, ln_b, b_o, wbf, out);
}
